// MultiScaleRetention_31636729102529
// MI455X (gfx1250) — matrix operand order repaired and checked by reading, not compiled, not run on hardware
//
#include <hip/hip_runtime.h>
#include <hip/hip_bf16.h>

#define B_    4
#define L_    2048
#define DIM_  1024
#define H_    8
#define HEAD_ 128
#define BL_   (B_ * L_)

typedef __bf16 bf16_t;
typedef __attribute__((ext_vector_type(16))) bf16_t v16bf;
typedef __attribute__((ext_vector_type(8)))  bf16_t v8bf;
typedef __attribute__((ext_vector_type(8)))  float  v8f;
typedef __attribute__((ext_vector_type(4)))  unsigned int v4u;
typedef __attribute__((ext_vector_type(4)))  int    v4i;
typedef __attribute__((ext_vector_type(8)))  int    v8i;

#if defined(__gfx1250__) && __has_builtin(__builtin_amdgcn_tensor_load_to_lds)
#define USE_TDM 1
#else
#define USE_TDM 0
#endif

// ---------- helpers ----------

__device__ __forceinline__ unsigned short f2bf(float f) {
  union { float f; unsigned u; } x; x.f = f;
  unsigned u = x.u;
  unsigned r = u + 0x7FFFu + ((u >> 16) & 1u);   // round-to-nearest-even
  if ((u & 0x7F800000u) == 0x7F800000u) r = u;   // inf/nan passthrough
  return (unsigned short)(r >> 16);
}

union ABfrag { v16bf v; v8bf h[2]; };

// A fragment (16x32 bf16, row-major source, rowStride in elements).
// A layout matching B: lane(half,row): elems 0..15 -> K = 16*half + 0..15
// (contiguous) => two adjacent 16B runs.
__device__ __forceinline__ v16bf load_A_frag(const unsigned short* base, int rowStride) {
  const int lane = threadIdx.x & 31;
  const int hl   = lane >> 4;
  const int row  = lane & 15;
  const unsigned short* p = base + (size_t)row * rowStride + 16 * hl;
  ABfrag f;
  f.h[0] = *(const v8bf*)(const void*)(p);
  f.h[1] = *(const v8bf*)(const void*)(p + 8);
  return f.v;
}

// B fragment (32x16 bf16) from a K-contiguous source:
// element (k,n) at base + n*colStride + k. Lane n loads K = 16*half + 0..15.
__device__ __forceinline__ v16bf load_B_frag(const unsigned short* base, int colStride) {
  const int lane = threadIdx.x & 31;
  const int hl   = lane >> 4;
  const int n    = lane & 15;
  return *(const v16bf*)(const void*)(base + (size_t)n * colStride + 16 * hl);
}

__device__ __forceinline__ v8f wmma_bf16(v16bf a, v16bf b, v8f c) {
  return __builtin_amdgcn_wmma_f32_16x16x32_bf16(false, a, false, b, (short)0, c, false, false);
}

#if USE_TDM
// Issue a TDM 2D tile load (global -> LDS). data_size = 2 bytes (bf16).
// D# Group0: [1:0]count=1, [63:32]lds_addr, [120:64]global_addr, [127:126]type=2
// D# Group1 packs data_size, tensor_dim0/1, tile_dim0/1, tensor_dim0_stride.
// Toolchain uses the 6-arg builtin: (v4u g0, v8i g1, v4i g2, v4i g3, v8i gx, i32 cpol)
__device__ __forceinline__ void tdm_load_2d(const void* gptr, unsigned lds_off,
                                            unsigned tensor_d0, unsigned tensor_d1,
                                            unsigned tile_d0, unsigned tile_d1,
                                            unsigned stride0) {
  unsigned long long ga = (unsigned long long)(size_t)gptr;
  v4u g0;
  g0[0] = 1u;                                   // count=1, user descriptor
  g0[1] = lds_off;                              // LDS byte address
  g0[2] = (unsigned)ga;                         // global_addr[31:0]
  g0[3] = (unsigned)((ga >> 32) & 0x01FFFFFFu)  // global_addr[56:32]
        | 0x80000000u;                          // type=2 ("image")
  v8i g1;
  g1[0] = (int)(1u << 16);                                    // data_size=1 -> 2B
  g1[1] = (int)(tensor_d0 << 16);                             // tensor_dim0[15:0]
  g1[2] = (int)((tensor_d0 >> 16) | (tensor_d1 << 16));       // td0[31:16] | td1[15:0]
  g1[3] = (int)((tensor_d1 >> 16) | (tile_d0 << 16));         // td1[31:16] | tile_dim0
  g1[4] = (int)(tile_d1 & 0xFFFFu);                           // tile_dim1 (tile_dim2=0)
  g1[5] = (int)stride0;                                       // tensor_dim0_stride[31:0]
  g1[6] = 0;                                                  // stride0[47:32]|stride1
  g1[7] = 0;
  v4i gz4 = {0, 0, 0, 0};                                     // groups 2/3: 2D only
  v8i gz8 = {0, 0, 0, 0, 0, 0, 0, 0};
  __builtin_amdgcn_tensor_load_to_lds(g0, g1, gz4, gz4, gz8, 0);
}
#endif

// ---------- prep kernels ----------

__global__ void k_f32_to_bf16(const float* __restrict__ src,
                              unsigned short* __restrict__ dst, long n) {
  for (long i = blockIdx.x * (long)blockDim.x + threadIdx.x; i < n;
       i += (long)gridDim.x * blockDim.x)
    dst[i] = f2bf(src[i]);
}

// dst[b][c][r] = bf16(src[b][r][c])  (K-contiguous weight layout)
__global__ void k_transpose_bf16(const float* __restrict__ src,
                                 unsigned short* __restrict__ dst,
                                 int batch, int rows, int cols) {
  long per = (long)rows * cols;
  long total = (long)batch * per;
  for (long i = blockIdx.x * (long)blockDim.x + threadIdx.x; i < total;
       i += (long)gridDim.x * blockDim.x) {
    long b  = i / per;
    long rc = i - b * per;
    long c  = rc / rows;
    long r  = rc - c * rows;
    dst[i] = f2bf(src[b * per + r * cols + c]);
  }
}

// ---------- QKV projection + xPos rotary ----------
// grid (BL/16, H), block 256 (8 waves); wave w owns e-tile [16w,16w+16)

__global__ void k_proj_xpos(const unsigned short* __restrict__ Xbf,
                            const unsigned short* __restrict__ WtQ,
                            const unsigned short* __restrict__ WtK,
                            const unsigned short* __restrict__ WtV,
                            unsigned short* __restrict__ Qb,
                            unsigned short* __restrict__ Kb,
                            unsigned short* __restrict__ Vt) {
  const int row0 = blockIdx.x * 16;
  const int h    = blockIdx.y;
  const int w    = threadIdx.x >> 5;
  const int lane = threadIdx.x & 31;
  const int hl   = lane >> 4;
  const int ln   = lane & 15;

  const unsigned short* wq = WtQ + ((size_t)h * HEAD_ + 16 * w) * DIM_;
  const unsigned short* wk = WtK + ((size_t)h * HEAD_ + 16 * w) * DIM_;
  const unsigned short* wv = WtV + ((size_t)h * HEAD_ + 16 * w) * DIM_;
  const unsigned short* xa = Xbf + (size_t)row0 * DIM_;

  v8f qacc = {}, kacc = {}, vacc = {};

  // software pipeline: prefetch chunk k0+32 fragments before chunk k0 WMMAs
  v16bf a_c  = load_A_frag(xa, DIM_);
  v16bf bq_c = load_B_frag(wq, DIM_);
  v16bf bk_c = load_B_frag(wk, DIM_);
  v16bf bv_c = load_B_frag(wv, DIM_);
  for (int k0 = 0; k0 < DIM_; k0 += 32) {
    const int kp = (k0 + 32 < DIM_) ? k0 + 32 : 0;
    v16bf a_n  = load_A_frag(xa + kp, DIM_);
    v16bf bq_n = load_B_frag(wq + kp, DIM_);
    v16bf bk_n = load_B_frag(wk + kp, DIM_);
    v16bf bv_n = load_B_frag(wv + kp, DIM_);
    qacc = wmma_bf16(a_c, bq_c, qacc);
    kacc = wmma_bf16(a_c, bk_c, kacc);
    vacc = wmma_bf16(a_c, bv_c, vacc);
    a_c = a_n; bq_c = bq_n; bk_c = bk_n; bv_c = bv_n;
  }

  const int   e        = 16 * w + ln;
  const int   jh       = e >> 1;
  const float inv_freq = __powf(10000.f, -(float)jh * (1.f / 64.f));
  const float basej    = (2.f * (float)jh + 0.4f * (float)HEAD_) / (1.4f * (float)HEAD_);

  #pragma unroll
  for (int j = 0; j < 8; ++j) {
    int r = row0 + j + 8 * hl;
    int b = r >> 11;             // r / L_
    int l = r & (L_ - 1);
    float ang = (float)l * inv_freq;
    float sn = __sinf(ang), cs = __cosf(ang);
    float sc  = __powf(basej, (float)l * (1.f / 512.f));
    float isc = 1.f / sc;

    float qv = qacc[j];
    float qp = __shfl_xor(qv, 1);
    float qo = (qv * cs + ((e & 1) ? qp : -qp) * sn) * sc;

    float kv = kacc[j];
    float kp2 = __shfl_xor(kv, 1);
    float ko = (kv * cs + ((e & 1) ? kp2 : -kp2) * sn) * isc;

    size_t bh = (size_t)b * H_ + h;
    Qb[(bh * L_ + l) * HEAD_ + e] = f2bf(qo);
    Kb[(bh * L_ + l) * HEAD_ + e] = f2bf(ko);
    Vt[(bh * HEAD_ + e) * L_ + l] = f2bf(vacc[j]);   // V stored transposed
  }
}

// ---------- retention (flash-style causal, decay mask) + groupnorm ----------
// grid (L/16, B*H), block 32 (one wave). TDM double-buffers K/V tiles in LDS.

__global__ void k_retention(const unsigned short* __restrict__ Qb,
                            const unsigned short* __restrict__ Kb,
                            const unsigned short* __restrict__ Vt,
                            const float* __restrict__ gnw,
                            const float* __restrict__ gnb,
                            float* __restrict__ Yn) {
  __shared__ __align__(32) unsigned short sS[16 * 32];     // C->A score relayout
#if USE_TDM
  __shared__ __align__(128) unsigned short sK[2][32 * HEAD_];   // 8KB x2
  __shared__ __align__(128) unsigned short sV[2][HEAD_ * 32];   // 8KB x2
#endif

  const int n0   = blockIdx.x * 16;
  const int bh   = blockIdx.y;
  const int h    = bh & (H_ - 1);
  const int b    = bh >> 3;
  const int lane = threadIdx.x & 31;
  const int hl   = lane >> 4;
  const int ln   = lane & 15;

  const float gamma = 1.f - __expf(-3.4657359f - 0.39608410f * (float)h);
  const float lg    = __logf(gamma);

  const unsigned short* qbase = Qb + ((size_t)bh * L_ + n0) * HEAD_;
  const unsigned short* kbase = Kb + (size_t)bh * L_ * HEAD_;
  const unsigned short* vbase = Vt + (size_t)bh * HEAD_ * L_;

  v16bf qa[4];
  #pragma unroll
  for (int ec = 0; ec < 4; ++ec) qa[ec] = load_A_frag(qbase + 32 * ec, HEAD_);

  v8f yacc[8] = {};

  const int nchunks = (n0 >> 5) + 1;

#if USE_TDM
  const unsigned ldsK0 = (unsigned)(size_t)(const void*)&sK[0][0];
  const unsigned ldsK1 = (unsigned)(size_t)(const void*)&sK[1][0];
  const unsigned ldsV0 = (unsigned)(size_t)(const void*)&sV[0][0];
  const unsigned ldsV1 = (unsigned)(size_t)(const void*)&sV[1][0];
  // K tile: rows m0..m0+31 of [L_ x HEAD_]; V tile: cols m0..m0+31 of [HEAD_ x L_]
  tdm_load_2d(kbase, ldsK0, HEAD_, L_, HEAD_, 32, HEAD_);
  tdm_load_2d(vbase, ldsV0, L_, HEAD_, 32, HEAD_, L_);
#endif

  for (int ci = 0; ci < nchunks; ++ci) {
    const int m0  = ci << 5;
    const int buf = ci & 1;

#if USE_TDM
    if (ci + 1 < nchunks) {
      const int m1 = m0 + 32;
      // DS reads of the other buffer finished last iteration; TDM is unordered
      // with DS, so drain DScnt before letting TDM overwrite that buffer.
      asm volatile("s_wait_dscnt 0x0" ::: "memory");
      tdm_load_2d(kbase + (size_t)m1 * HEAD_, buf ? ldsK0 : ldsK1,
                  HEAD_, L_, HEAD_, 32, HEAD_);
      tdm_load_2d(vbase + m1, buf ? ldsV0 : ldsV1,
                  L_, HEAD_, 32, HEAD_, L_);
      __builtin_amdgcn_s_wait_tensorcnt((short)2);  // current chunk's 2 loads done
    } else {
      __builtin_amdgcn_s_wait_tensorcnt((short)0);
    }
    const unsigned short* kt = &sK[buf][0];   // [32][HEAD_]
    const unsigned short* vt = &sV[buf][0];   // [HEAD_][32]
    const int kStride = HEAD_, vStride = 32;
    const size_t vTile = 32;
#else
    const unsigned short* kt = kbase + (size_t)m0 * HEAD_;
    const unsigned short* vt = vbase + m0;
    const int kStride = HEAD_, vStride = L_;
    const size_t vTile = L_;
#endif

    v8f s0 = {}, s1 = {};
    #pragma unroll
    for (int ec = 0; ec < 4; ++ec) {
      s0 = wmma_bf16(qa[ec], load_B_frag(kt + 32 * ec, kStride), s0);
      s1 = wmma_bf16(qa[ec], load_B_frag(kt + (size_t)16 * kStride + 32 * ec, kStride), s1);
    }

    // elementwise decay mask, stage score tile row-major in LDS (C->A relayout)
    #pragma unroll
    for (int j = 0; j < 8; ++j) {
      int row = n0 + j + 8 * hl;
      int c0  = m0 + ln;
      int c1  = m0 + 16 + ln;
      float v0 = (row >= c0) ? s0[j] * __expf((float)(row - c0) * lg) : 0.f;
      float v1 = (row >= c1) ? s1[j] * __expf((float)(row - c1) * lg) : 0.f;
      sS[(j + 8 * hl) * 32 + ln]      = f2bf(v0);
      sS[(j + 8 * hl) * 32 + 16 + ln] = f2bf(v1);
    }
    // LDS ops are in-order within a wave: safe to reload as A fragment
    v16bf sa = load_A_frag(sS, 32);

    #pragma unroll
    for (int c = 0; c < 8; ++c)
      yacc[c] = wmma_bf16(sa, load_B_frag(vt + (size_t)(16 * c) * vTile, vStride), yacc[c]);
  }

  // groupnorm per row over HEAD_=128 (rows live per 16-lane half)
  #pragma unroll
  for (int j = 0; j < 8; ++j) {
    float s = 0.f, s2 = 0.f;
    #pragma unroll
    for (int c = 0; c < 8; ++c) { float y = yacc[c][j]; s += y; s2 += y * y; }
    #pragma unroll
    for (int o = 1; o < 16; o <<= 1) { s += __shfl_xor(s, o); s2 += __shfl_xor(s2, o); }
    float mean = s  * (1.f / HEAD_);
    float var  = s2 * (1.f / HEAD_) - mean * mean;
    float rstd = rsqrtf(var + 1e-5f);
    int l = n0 + j + 8 * hl;
    #pragma unroll
    for (int c = 0; c < 8; ++c) {
      int col = 16 * c + ln;
      float yn = (yacc[c][j] - mean) * rstd;
      yn = yn * gnw[h * HEAD_ + col] + gnb[h * HEAD_ + col];
      Yn[((size_t)b * L_ + l) * DIM_ + h * HEAD_ + col] = yn;
    }
  }
}

// ---------- gate: Z = silu(X @ W_G) * Yn ----------
// grid (BL/16, DIM/64), block 32: one wave, 4 col-tiles sharing A fragments

__global__ void k_gate_silu(const unsigned short* __restrict__ Xbf,
                            const unsigned short* __restrict__ WtG,
                            const float* __restrict__ Yn,
                            unsigned short* __restrict__ Z) {
  const int row0 = blockIdx.x * 16;
  const int col0 = blockIdx.y * 64;
  const int lane = threadIdx.x & 31;
  const int hl   = lane >> 4;
  const int ln   = lane & 15;

  v8f acc[4] = {};
  const unsigned short* xa = Xbf + (size_t)row0 * DIM_;

  v16bf a_c = load_A_frag(xa, DIM_);
  v16bf b_c[4];
  #pragma unroll
  for (int t = 0; t < 4; ++t)
    b_c[t] = load_B_frag(WtG + (size_t)(col0 + 16 * t) * DIM_, DIM_);

  for (int k0 = 0; k0 < DIM_; k0 += 32) {
    const int kp = (k0 + 32 < DIM_) ? k0 + 32 : 0;
    v16bf a_n = load_A_frag(xa + kp, DIM_);
    v16bf b_n[4];
    #pragma unroll
    for (int t = 0; t < 4; ++t)
      b_n[t] = load_B_frag(WtG + (size_t)(col0 + 16 * t) * DIM_ + kp, DIM_);
    #pragma unroll
    for (int t = 0; t < 4; ++t) acc[t] = wmma_bf16(a_c, b_c[t], acc[t]);
    a_c = a_n;
    #pragma unroll
    for (int t = 0; t < 4; ++t) b_c[t] = b_n[t];
  }

  #pragma unroll
  for (int t = 0; t < 4; ++t) {
    #pragma unroll
    for (int j = 0; j < 8; ++j) {
      int row = row0 + j + 8 * hl;
      int col = col0 + 16 * t + ln;
      float g = acc[t][j];
      float sig = 1.f / (1.f + __expf(-g));
      float z = g * sig * Yn[(size_t)row * DIM_ + col];
      Z[(size_t)row * DIM_ + col] = f2bf(z);
    }
  }
}

// ---------- out = Z @ W_O ----------

__global__ void k_out_proj(const unsigned short* __restrict__ Z,
                           const unsigned short* __restrict__ WtO,
                           float* __restrict__ out) {
  const int row0 = blockIdx.x * 16;
  const int col0 = blockIdx.y * 64;
  const int lane = threadIdx.x & 31;
  const int hl   = lane >> 4;
  const int ln   = lane & 15;

  v8f acc[4] = {};
  const unsigned short* za = Z + (size_t)row0 * DIM_;

  v16bf a_c = load_A_frag(za, DIM_);
  v16bf b_c[4];
  #pragma unroll
  for (int t = 0; t < 4; ++t)
    b_c[t] = load_B_frag(WtO + (size_t)(col0 + 16 * t) * DIM_, DIM_);

  for (int k0 = 0; k0 < DIM_; k0 += 32) {
    const int kp = (k0 + 32 < DIM_) ? k0 + 32 : 0;
    v16bf a_n = load_A_frag(za + kp, DIM_);
    v16bf b_n[4];
    #pragma unroll
    for (int t = 0; t < 4; ++t)
      b_n[t] = load_B_frag(WtO + (size_t)(col0 + 16 * t) * DIM_ + kp, DIM_);
    #pragma unroll
    for (int t = 0; t < 4; ++t) acc[t] = wmma_bf16(a_c, b_c[t], acc[t]);
    a_c = a_n;
    #pragma unroll
    for (int t = 0; t < 4; ++t) b_c[t] = b_n[t];
  }

  #pragma unroll
  for (int t = 0; t < 4; ++t) {
    #pragma unroll
    for (int j = 0; j < 8; ++j) {
      int row = row0 + j + 8 * hl;
      int col = col0 + 16 * t + ln;
      out[(size_t)row * DIM_ + col] = acc[t][j];
    }
  }
}

// ---------- host launcher ----------

extern "C" void kernel_launch(void* const* d_in, const int* in_sizes, int n_in,
                              void* d_out, int out_size, void* d_ws, size_t ws_size,
                              hipStream_t stream) {
  (void)in_sizes; (void)n_in; (void)out_size; (void)ws_size;

  const float* X   = (const float*)d_in[0];
  const float* W_Q = (const float*)d_in[1];
  const float* W_K = (const float*)d_in[2];
  const float* W_V = (const float*)d_in[3];
  const float* W_G = (const float*)d_in[4];
  const float* W_O = (const float*)d_in[5];
  const float* gnw = (const float*)d_in[6];
  const float* gnb = (const float*)d_in[7];

  char* ws = (char*)d_ws;
  size_t off = 0;
  auto carve = [&](size_t bytes) -> void* {
    void* p = ws + off;
    off += (bytes + 255) & ~(size_t)255;
    return p;
  };

  unsigned short* Xbf = (unsigned short*)carve((size_t)BL_ * DIM_ * 2);
  unsigned short* WtQ = (unsigned short*)carve((size_t)H_ * HEAD_ * DIM_ * 2);
  unsigned short* WtK = (unsigned short*)carve((size_t)H_ * HEAD_ * DIM_ * 2);
  unsigned short* WtV = (unsigned short*)carve((size_t)H_ * HEAD_ * DIM_ * 2);
  unsigned short* WtG = (unsigned short*)carve((size_t)DIM_ * DIM_ * 2);
  unsigned short* WtO = (unsigned short*)carve((size_t)DIM_ * DIM_ * 2);
  unsigned short* Qb  = (unsigned short*)carve((size_t)B_ * H_ * L_ * HEAD_ * 2);
  unsigned short* Kb  = (unsigned short*)carve((size_t)B_ * H_ * L_ * HEAD_ * 2);
  unsigned short* Vt  = (unsigned short*)carve((size_t)B_ * H_ * L_ * HEAD_ * 2);
  float*          Yn  = (float*)carve((size_t)BL_ * DIM_ * 4);
  unsigned short* Z   = (unsigned short*)carve((size_t)BL_ * DIM_ * 2);

  k_f32_to_bf16<<<dim3(2048), dim3(256), 0, stream>>>(X, Xbf, (long)BL_ * DIM_);
  k_transpose_bf16<<<dim3(1024), dim3(256), 0, stream>>>(W_Q, WtQ, H_, DIM_, HEAD_);
  k_transpose_bf16<<<dim3(1024), dim3(256), 0, stream>>>(W_K, WtK, H_, DIM_, HEAD_);
  k_transpose_bf16<<<dim3(1024), dim3(256), 0, stream>>>(W_V, WtV, H_, DIM_, HEAD_);
  k_transpose_bf16<<<dim3(1024), dim3(256), 0, stream>>>(W_G, WtG, 1, DIM_, DIM_);
  k_transpose_bf16<<<dim3(1024), dim3(256), 0, stream>>>(W_O, WtO, 1, DIM_, DIM_);

  k_proj_xpos<<<dim3(BL_ / 16, H_), dim3(256), 0, stream>>>(Xbf, WtQ, WtK, WtV, Qb, Kb, Vt);
  k_retention<<<dim3(L_ / 16, B_ * H_), dim3(32), 0, stream>>>(Qb, Kb, Vt, gnw, gnb, Yn);
  k_gate_silu<<<dim3(BL_ / 16, DIM_ / 64), dim3(32), 0, stream>>>(Xbf, WtG, Yn, Z);
  k_out_proj<<<dim3(BL_ / 16, DIM_ / 64), dim3(32), 0, stream>>>(Z, WtO, (float*)d_out);
}
